// Actor_87497073754359
// MI455X (gfx1250) — matrix operand order repaired and checked by reading, not compiled, not run on hardware
//
#include <hip/hip_runtime.h>
#include <hip/hip_bf16.h>

#define B_ 128
#define N_ 2048
#define D_ 128
#define H_ 8

typedef __attribute__((ext_vector_type(16))) __bf16 v16bf;
typedef __attribute__((ext_vector_type(8)))  float  v8f;

#if defined(__has_builtin)
#if __has_builtin(__builtin_amdgcn_sched_barrier)
#define SCHED_FENCE() __builtin_amdgcn_sched_barrier(0)
#endif
#endif
#ifndef SCHED_FENCE
#define SCHED_FENCE()
#endif

__device__ __forceinline__ unsigned short f2bf_rne(float f) {
  unsigned int u = __float_as_uint(f);
  u += 0x7FFFu + ((u >> 16) & 1u);   // round-to-nearest-even truncation to bf16
  return (unsigned short)(u >> 16);
}

// ---- Kernel 1: graph_emb[b][c] = mean_n x[b][n][c]; optionally emit x as bf16 ----
__global__ __launch_bounds__(256) void k_mean_bf(const float* __restrict__ x,
                                                 float* __restrict__ ge,
                                                 unsigned int* __restrict__ xbf, // 2 bf16/word
                                                 int write_bf) {
  __shared__ float p[256 * 4];
  const int b = blockIdx.x, t = threadIdx.x;
  const int c0 = (t & 31) * 4;     // 4 columns per lane -> 32 lanes cover all 128
  const int g  = t >> 5;           // 8 row groups
  const float4* xb = (const float4*)(x + (size_t)b * N_ * D_);
  unsigned int* xo = xbf + ((size_t)b * N_ * D_) / 2;
  float s0 = 0.f, s1 = 0.f, s2 = 0.f, s3 = 0.f;
  for (int n = g; n < N_; n += 8) {
    const float4 v = xb[(size_t)n * (D_ / 4) + (c0 >> 2)];
    s0 += v.x; s1 += v.y; s2 += v.z; s3 += v.w;
    if (write_bf) {
      uint2 pk;
      pk.x = (unsigned int)f2bf_rne(v.x) | ((unsigned int)f2bf_rne(v.y) << 16);
      pk.y = (unsigned int)f2bf_rne(v.z) | ((unsigned int)f2bf_rne(v.w) << 16);
      *(uint2*)(xo + (size_t)n * (D_ / 2) + (c0 >> 1)) = pk;
    }
  }
  p[t * 4 + 0] = s0; p[t * 4 + 1] = s1; p[t * 4 + 2] = s2; p[t * 4 + 3] = s3;
  __syncthreads();
  if (t < 128) {
    const int c = t;
    float a = 0.f;
#pragma unroll
    for (int gg = 0; gg < 8; ++gg) a += p[(gg * 32 + (c >> 2)) * 4 + (c & 3)];
    ge[b * D_ + c] = a * (1.0f / N_);
  }
}

// ---------- Kernel 2: context -> q_in -> q -> Weff/biasEff -> bf16 B-image ----------
__global__ __launch_bounds__(128) void k_prep(const float* __restrict__ x,
                                              const int* __restrict__ fn,
                                              const int* __restrict__ cn,
                                              const float* __restrict__ ge,
                                              const float* __restrict__ W_lin,
                                              const float* __restrict__ b_lin,
                                              const float* __restrict__ Wq,
                                              const float* __restrict__ bq,
                                              const float* __restrict__ Wk,
                                              const float* __restrict__ bk,
                                              float* __restrict__ biasEff,
                                              unsigned short* __restrict__ bimg) {
  __shared__ float ctx[384], qin[128], qv[128], weff[H_ * 128];
  const int b = blockIdx.x, t = threadIdx.x;
  const int i_f = fn[b], i_c = cn[b];
  ctx[t]       = ge[b * D_ + t];
  ctx[128 + t] = x[((size_t)b * N_ + i_f) * D_ + t];
  ctx[256 + t] = x[((size_t)b * N_ + i_c) * D_ + t];
  __syncthreads();
  {
    float a = b_lin[t];
    const float* w = W_lin + (size_t)t * 384;   // q_in[d] = ctx . W_lin[d,:]
    for (int k = 0; k < 384; ++k) a += ctx[k] * w[k];
    qin[t] = a;
  }
  __syncthreads();
  {
    float a = bq[t];
    const float* w = Wq + (size_t)t * 128;      // q[d] = q_in . Wq[d,:]
    for (int k = 0; k < 128; ++k) a += qin[k] * w[k];
    qv[t] = a;
  }
  __syncthreads();
  // Weff[h][c] = 0.25 * sum_j qv[16h+j] * Wk[16h+j, c]   (0.25 = 1/sqrt(HD))
  for (int h = 0; h < H_; ++h) {
    float a = 0.f;
    for (int j = 0; j < 16; ++j) a += qv[h * 16 + j] * Wk[(size_t)(h * 16 + j) * 128 + t];
    weff[h * 128 + t] = 0.25f * a;
  }
  if (t < H_) {
    float a = 0.f;
    for (int j = 0; j < 16; ++j) a += qv[t * 16 + j] * bk[t * 16 + j];
    biasEff[b * H_ + t] = 0.25f * a;
  }
  __syncthreads();
  // Pre-swizzle WMMA B operand: [chunk kk][lane: head col = lane%16][elem: K in table order]
  for (int idx = t; idx < 4 * 32 * 16; idx += 128) {
    const int kk = idx >> 9, rem = idx & 511, lane = rem >> 4, e = rem & 15;
    const int c = kk * 32 + ((e < 8) ? (8 * (lane >> 4) + e) : (16 + 8 * (lane >> 4) + (e - 8)));
    const float v = ((lane & 15) < H_) ? weff[(lane & 15) * 128 + c] : 0.f;
    bimg[(size_t)b * 2048 + idx] = f2bf_rne(v);
  }
}

// ----- Kernel 3 (fast path): scores from pre-converted bf16 x; pipelined A loads -----
__global__ __launch_bounds__(256) void k_scores_bf(const unsigned short* __restrict__ xbf,
                                                   const int* __restrict__ mask,
                                                   const float* __restrict__ biasEff,
                                                   const unsigned short* __restrict__ bimg,
                                                   float* __restrict__ sc) {
  const int b    = blockIdx.x;
  const int lane = threadIdx.x & 31;
  const int wave = threadIdx.x >> 5;
  const int slot = blockIdx.y * 8 + wave;          // 0..15 tile slots per batch
  const v16bf* bp = (const v16bf*)(bimg + (size_t)b * 2048);
  const v16bf b0 = bp[0 * 32 + lane];
  const v16bf b1 = bp[1 * 32 + lane];
  const v16bf b2 = bp[2 * 32 + lane];
  const v16bf b3 = bp[3 * 32 + lane];
  const int row_in_tile = lane & 15;
  const int koff        = (lane >> 4);             // kbase = 8*koff; uint4 units: koff
  const int h           = lane & 15;
  const float bias = (h < H_) ? biasEff[b * H_ + h] : 0.f;
  const unsigned short* xrow0 = xbf + ((size_t)b * N_ + row_in_tile) * D_;

  for (int i = 0; i < 8; ++i) {
    const int tile = slot + 16 * i;
    const int n0   = tile * 16;
    const uint4* xr = (const uint4*)(xrow0 + (size_t)n0 * D_);
    const int rbase = n0 + koff * 8;               // C/D: VGPR r -> row r + 8*(lane>>4)
    // Prefetch next tile's A rows (256 rows ahead) -> global_prefetch_b8
    if (i < 7) __builtin_prefetch((const void*)(xrow0 + (size_t)(n0 + 256) * D_), 0, 3);
    // Hoist mask loads so their latency hides under the WMMA chain.
    int4 m0 = {0, 0, 0, 0}, m1 = {0, 0, 0, 0};
    if (h < H_) {
      const int4* mp = (const int4*)(mask + (size_t)b * N_ + rbase);
      m0 = mp[0];
      m1 = mp[1];
    }
    // Issue all 8 A loads; fence keeps them clustered ahead of the WMMA chain
    // so waits are staggered (loadcnt 6/4/2/0) instead of full serialization.
    union Au { uint4 q[2]; v16bf v; } a0, a1, a2, a3;
    a0.q[0] = xr[0  + koff];  a0.q[1] = xr[2  + koff];
    a1.q[0] = xr[4  + koff];  a1.q[1] = xr[6  + koff];
    a2.q[0] = xr[8  + koff];  a2.q[1] = xr[10 + koff];
    a3.q[0] = xr[12 + koff];  a3.q[1] = xr[14 + koff];
    SCHED_FENCE();
    v8f acc = {};
    acc = __builtin_amdgcn_wmma_f32_16x16x32_bf16(false, a0.v, false, b0, (short)0, acc, false, false);
    acc = __builtin_amdgcn_wmma_f32_16x16x32_bf16(false, a1.v, false, b1, (short)0, acc, false, false);
    acc = __builtin_amdgcn_wmma_f32_16x16x32_bf16(false, a2.v, false, b2, (short)0, acc, false, false);
    acc = __builtin_amdgcn_wmma_f32_16x16x32_bf16(false, a3.v, false, b3, (short)0, acc, false, false);
    if (h < H_) {
      float* so = sc + ((size_t)b * H_ + h) * N_;
      float v0 = acc[0] + bias, v1 = acc[1] + bias, v2 = acc[2] + bias, v3 = acc[3] + bias;
      float v4 = acc[4] + bias, v5 = acc[5] + bias, v6 = acc[6] + bias, v7 = acc[7] + bias;
      const float ninf = -__builtin_inff();
      if (m0.x) v0 = ninf;  if (m0.y) v1 = ninf;  if (m0.z) v2 = ninf;  if (m0.w) v3 = ninf;
      if (m1.x) v4 = ninf;  if (m1.y) v5 = ninf;  if (m1.z) v6 = ninf;  if (m1.w) v7 = ninf;
      so[rbase + 0] = v0;  so[rbase + 1] = v1;  so[rbase + 2] = v2;  so[rbase + 3] = v3;
      so[rbase + 4] = v4;  so[rbase + 5] = v5;  so[rbase + 6] = v6;  so[rbase + 7] = v7;
    }
  }
}

// ----- Kernel 3 (fallback): convert f32 x -> bf16 on the fly (small-workspace path) -----
__global__ __launch_bounds__(256) void k_scores_f32(const float* __restrict__ x,
                                                    const int* __restrict__ mask,
                                                    const float* __restrict__ biasEff,
                                                    const unsigned short* __restrict__ bimg,
                                                    float* __restrict__ sc) {
  const int b    = blockIdx.x;
  const int lane = threadIdx.x & 31;
  const int wave = threadIdx.x >> 5;
  const int slot = blockIdx.y * 8 + wave;
  const v16bf* bp = (const v16bf*)(bimg + (size_t)b * 2048);
  const v16bf b0 = bp[0 * 32 + lane];
  const v16bf b1 = bp[1 * 32 + lane];
  const v16bf b2 = bp[2 * 32 + lane];
  const v16bf b3 = bp[3 * 32 + lane];
  const int row_in_tile = lane & 15;
  const int kbase       = (lane >> 4) * 8;
  const int h           = lane & 15;
  const float bias = (h < H_) ? biasEff[b * H_ + h] : 0.f;
  const float* xb = x + (size_t)b * N_ * D_;

  for (int i = 0; i < 8; ++i) {
    const int tile = slot + 16 * i;
    const int n0   = tile * 16;
    const float* xr = xb + (size_t)(n0 + row_in_tile) * D_;
    v8f acc = {};
    union { unsigned short u[16]; v16bf v; } a;
#pragma unroll
    for (int kk = 0; kk < 4; ++kk) {
      const float* p0 = xr + kk * 32 + kbase;
      const float* p1 = p0 + 16;
#pragma unroll
      for (int e = 0; e < 8; ++e) {
        a.u[e]     = f2bf_rne(p0[e]);
        a.u[e + 8] = f2bf_rne(p1[e]);
      }
      const v16bf bm = (kk == 0) ? b0 : (kk == 1) ? b1 : (kk == 2) ? b2 : b3;
      acc = __builtin_amdgcn_wmma_f32_16x16x32_bf16(false, a.v, false, bm,
                                                    (short)0, acc, false, false);
    }
    if (h < H_) {
      float* so = sc + ((size_t)b * H_ + h) * N_;
      const int rbase = n0 + ((lane >> 4) * 8);
#pragma unroll
      for (int r = 0; r < 8; ++r) {
        const int n = rbase + r;
        float v = acc[r] + bias;
        if (mask[(size_t)b * N_ + n]) v = -__builtin_inff();
        so[n] = v;
      }
    }
  }
}

// ------------- Kernel 4: masked softmax over N per head + mean over heads -------------
__global__ __launch_bounds__(256) void k_softmax_mean(const float* __restrict__ sc,
                                                      float* __restrict__ out) {
  __shared__ float hmax[H_], hinv[H_];
  const int b    = blockIdx.x;
  const int lane = threadIdx.x & 31;
  const int h    = threadIdx.x >> 5;               // wave32 == one head
  const float* s = sc + ((size_t)b * H_ + h) * N_;
  float m = -__builtin_inff();
  for (int n = lane; n < N_; n += 32) m = fmaxf(m, s[n]);
  for (int o = 16; o > 0; o >>= 1) m = fmaxf(m, __shfl_xor(m, o, 32));
  float sum = 0.f;
  for (int n = lane; n < N_; n += 32) sum += __expf(s[n] - m);
  for (int o = 16; o > 0; o >>= 1) sum += __shfl_xor(sum, o, 32);
  if (lane == 0) { hmax[h] = m; hinv[h] = 1.0f / sum; }
  __syncthreads();
  float lm[H_], li[H_];
#pragma unroll
  for (int hh = 0; hh < H_; ++hh) { lm[hh] = hmax[hh]; li[hh] = hinv[hh]; }
  for (int n = threadIdx.x; n < N_; n += 256) {
    float a = 0.f;
#pragma unroll
    for (int hh = 0; hh < H_; ++hh)
      a += __expf(sc[((size_t)b * H_ + hh) * N_ + n] - lm[hh]) * li[hh];
    out[(size_t)b * N_ + n] = a * 0.125f;
  }
}

extern "C" void kernel_launch(void* const* d_in, const int* in_sizes, int n_in,
                              void* d_out, int out_size, void* d_ws, size_t ws_size,
                              hipStream_t stream) {
  (void)in_sizes; (void)n_in; (void)out_size;
  const float* x     = (const float*)d_in[0];
  const int*   fn    = (const int*)d_in[1];
  const int*   cn    = (const int*)d_in[2];
  const int*   mask  = (const int*)d_in[3];
  const float* W_lin = (const float*)d_in[4];
  const float* b_lin = (const float*)d_in[5];
  const float* Wq    = (const float*)d_in[6];
  const float* bq    = (const float*)d_in[7];
  const float* Wk    = (const float*)d_in[8];
  const float* bk    = (const float*)d_in[9];
  float* out = (float*)d_out;

  // Workspace: ge[B*D] | biasEff[B*H] | bimg(B*2048 u16) | sc[B*H*N] | xbf(B*N*D u16)
  float* ws = (float*)d_ws;
  float* ge      = ws;
  float* biasEff = ws + (size_t)B_ * D_;
  unsigned short* bimg = (unsigned short*)(ws + (size_t)B_ * D_ + (size_t)B_ * H_);
  float* sc = (float*)(bimg + (size_t)B_ * 2048);
  unsigned short* xbf = (unsigned short*)(sc + (size_t)B_ * H_ * N_);
  const size_t need_big = (size_t)((char*)(xbf + (size_t)B_ * N_ * D_) - (char*)d_ws);
  const int big = (ws_size >= need_big) ? 1 : 0;   // constant across calls -> deterministic

  k_mean_bf<<<B_, 256, 0, stream>>>(x, ge, (unsigned int*)xbf, big);
  k_prep<<<B_, 128, 0, stream>>>(x, fn, cn, ge, W_lin, b_lin, Wq, bq, Wk, bk, biasEff, bimg);
  if (big)
    k_scores_bf<<<dim3(B_, 2), 256, 0, stream>>>(xbf, mask, biasEff, bimg, sc);
  else
    k_scores_f32<<<dim3(B_, 2), 256, 0, stream>>>(x, mask, biasEff, bimg, sc);
  k_softmax_mean<<<B_, 256, 0, stream>>>(sc, out);
}
